// unit_gcn_72353019068845
// MI455X (gfx1250) — matrix operand order repaired and checked by reading, not compiled, not run on hardware
//
#include <hip/hip_runtime.h>
#include <hip/hip_bf16.h>

typedef __attribute__((ext_vector_type(16))) __bf16      v16bf;
typedef __attribute__((ext_vector_type(8)))  float        v8f;
typedef __attribute__((ext_vector_type(8)))  unsigned int v8u;

union FragB16 { v8u u; v16bf b; };

#define C_DIM   64
#define V_DIM   25
#define T_DIM   300
#define S_DIM   3
#define TCHUNK  16
#define NCHUNK  19                 // ceil(300/16)
#define JCOLS   (TCHUNK * V_DIM)   // 400

// LDS strides (bf16/ushort elements): rows 16B-aligned, bank-conflict-free,
// chosen so every WMMA fragment is two contiguous 16B runs (ds_load_b128).
#define XS_STRIDE 520              // 1040 B/row (260 dw; 4n mod 64 distinct)
#define AE_STRIDE 40               // 80 B/row  (20 dw; 20n mod 64 distinct)
#define WS_STRIDE 72               // 144 B/row (36 dw; 36n mod 64 distinct)
#define XA_STRIDE 72

// LDS partition (ushort units)
#define OFF_XS   0
#define OFF_AET  (OFF_XS  + C_DIM * XS_STRIDE)   // 33280
#define OFF_WS   (OFF_AET + 32 * AE_STRIDE)      // 34560
#define OFF_XAS  (OFF_WS  + C_DIM * WS_STRIDE)   // 39168
#define OFF_SC   (OFF_XAS + JCOLS * XA_STRIDE)   // 67968
#define SMEM_USHORTS (OFF_SC + 256 + 1280)       // sc/sh (128 f32) + OOB slack
#define SMEM_BYTES   (SMEM_USHORTS * 2)          // 139008 B -> 2 WGs/WGP

__device__ __forceinline__ unsigned int f32_to_bf16_rne(float f) {
    unsigned int x = __builtin_bit_cast(unsigned int, f);
    return ((x + 0x7FFFu + ((x >> 16) & 1u)) >> 16) & 0xFFFFu;
}
__device__ __forceinline__ unsigned int pack_bf16x2(float lo, float hi) {
    return f32_to_bf16_rne(lo) | (f32_to_bf16_rne(hi) << 16);
}

// A fragment: 16xK slice, element (m,k) at base[m*stride + k].
// ISA 16-bit A layout = two contiguous 16B runs per lane -> 2x ds_load_b128.
__device__ __forceinline__ v16bf load_afrag(const unsigned short* base, int stride, int lane) {
    const int m = lane & 15, half = lane >> 4;
    const unsigned short* p = base + m * stride + half * 8;
    uint4 q0 = *reinterpret_cast<const uint4*>(p);        // k = half*8 + 0..7
    uint4 q1 = *reinterpret_cast<const uint4*>(p + 16);   // k = 16 + half*8 + 0..7
    FragB16 f;
    f.u[0] = q0.x; f.u[1] = q0.y; f.u[2] = q0.z; f.u[3] = q0.w;
    f.u[4] = q1.x; f.u[5] = q1.y; f.u[6] = q1.z; f.u[7] = q1.w;
    return f.b;
}

// B fragment: Kx16 tile stored TRANSPOSED: element (k,n) at base[n*stride + k].
// Per lane: k = half*8 + 0..7 and 16 + half*8 + 0..7 -> 2x ds_load_b128.
__device__ __forceinline__ v16bf load_bfrag(const unsigned short* base, int stride, int lane) {
    const int n = lane & 15, half = lane >> 4;
    const unsigned short* p = base + n * stride + half * 8;
    uint4 q0 = *reinterpret_cast<const uint4*>(p);        // k = half*8 + 0..7
    uint4 q1 = *reinterpret_cast<const uint4*>(p + 16);   // k = 16 + half*8 + 0..7
    FragB16 f;
    f.u[0] = q0.x; f.u[1] = q0.y; f.u[2] = q0.z; f.u[3] = q0.w;
    f.u[4] = q1.x; f.u[5] = q1.y; f.u[6] = q1.z; f.u[7] = q1.w;
    return f.b;
}

extern "C" __global__ __launch_bounds__(256)
void unit_gcn_fused_wmma(const float* __restrict__ x,
                         const float* __restrict__ A,
                         const float* __restrict__ PA1,
                         const float* __restrict__ PA2,
                         const float* __restrict__ Wc,
                         const float* __restrict__ bc,
                         const float* __restrict__ gamma,
                         const float* __restrict__ beta,
                         const float* __restrict__ rmean,
                         const float* __restrict__ rvar,
                         float* __restrict__ out) {
    extern __shared__ unsigned short smem[];
    unsigned short* Xs  = smem + OFF_XS;
    unsigned short* AeT = smem + OFF_AET;
    unsigned short* Ws  = smem + OFF_WS;
    unsigned short* XAs = smem + OFF_XAS;
    float* sc = reinterpret_cast<float*>(smem + OFF_SC);
    float* sh = sc + 64;

    const int tid   = threadIdx.x;
    const int lane  = tid & 31;
    const int wv    = tid >> 5;          // wave 0..7
    const int chunk = blockIdx.x;        // 0..18
    const int n     = blockIdx.y;        // 0..63
    const int t0    = chunk * TCHUNK;
    const int tc    = (T_DIM - t0 < TCHUNK) ? (T_DIM - t0) : TCHUNK;
    const int jmax  = tc * V_DIM;

    // ---- per-channel BN scale/shift with summed bias folded in ----
    if (tid < 64) {
        float s0   = gamma[tid] * __frsqrt_rn(rvar[tid] + 1e-5f);
        float btot = bc[tid] + bc[64 + tid] + bc[128 + tid];
        sc[tid] = s0;
        sh[tid] = beta[tid] + (btot - rmean[tid]) * s0;
    }

    // ---- zero-fill Xs (covers V->32 / T->16 padding) with b128 stores ----
    {
        uint4 z; z.x = z.y = z.z = z.w = 0u;
        uint4* xs4 = reinterpret_cast<uint4*>(Xs);
        for (int i = tid; i < C_DIM * XS_STRIDE / 8; i += 256) xs4[i] = z;
    }
    __syncthreads();

    // ---- coalesced x chunk -> LDS bf16 (flat (t,v) range is contiguous) ----
    const float* xn = x + (size_t)n * C_DIM * T_DIM * V_DIM;
    {
        const int nflat = tc * V_DIM;
        for (int idx = tid; idx < C_DIM * JCOLS; idx += 256) {
            int c = idx / JCOLS;
            int q = idx - c * JCOLS;
            if (q < nflat) {
                float val = xn[c * (T_DIM * V_DIM) + t0 * V_DIM + q];
                int t = q / V_DIM, v = q - t * V_DIM;
                Xs[c * XS_STRIDE + t * 32 + v] = (unsigned short)f32_to_bf16_rne(val);
            }
        }
    }

    // stage-B per-wave tiles: o-tile = wv&3, 13 j-tiles per half of waves
    const int ow    = wv & 3;
    const int jg    = wv >> 2;
    const int jbase = jg * 13;           // jg==1 tile #12 is j>=400: dead tile

    v8f acc[13];
#pragma unroll
    for (int i = 0; i < 13; ++i) acc[i] = {};

    for (int s = 0; s < S_DIM; ++s) {
        __syncthreads();   // prev stage B finished; Xs ready on first iter

        // Ae = A*PA1 + PA2, stored transposed+padded: AeT[w][v], 32x32 bf16
        for (int idx = tid; idx < 32 * 32; idx += 256) {
            int w = idx >> 5, v = idx & 31;
            float val = 0.f;
            if (v < V_DIM && w < V_DIM) {
                int g = s * V_DIM * V_DIM + v * V_DIM + w;
                val = A[g] * PA1[g] + PA2[g];
            }
            AeT[w * AE_STRIDE + v] = (unsigned short)f32_to_bf16_rne(val);
        }
        // Wc[s] -> bf16 LDS
        for (int idx = tid; idx < 64 * 64; idx += 256)
            Ws[(idx >> 6) * WS_STRIDE + (idx & 63)] =
                (unsigned short)f32_to_bf16_rne(Wc[s * 4096 + idx]);
        __syncthreads();

        // ---- stage A: XA = X @ Ae[s]  (per frame: 64x32 x 32x25) ----
        v16bf bw0 = load_bfrag(AeT + 0  * AE_STRIDE, AE_STRIDE, lane);
        v16bf bw1 = load_bfrag(AeT + 16 * AE_STRIDE, AE_STRIDE, lane);
        const int half = lane >> 4, ncol = lane & 15;
#pragma unroll
        for (int tt = 0; tt < 2; ++tt) {
            int t = wv * 2 + tt;
#pragma unroll
            for (int ct = 0; ct < 4; ++ct) {
                v16bf a = load_afrag(Xs + (ct * 16) * XS_STRIDE + t * 32, XS_STRIDE, lane);
                v8f d0 = {}, d1 = {};
                d0 = __builtin_amdgcn_wmma_f32_16x16x32_bf16(false, a, false, bw0,
                                                             (short)0, d0, false, false);
                d1 = __builtin_amdgcn_wmma_f32_16x16x32_bf16(false, a, false, bw1,
                                                             (short)0, d1, false, false);
                // store transposed into XAs[j][c], j = t*25 + w, only w < 25
                {
                    int j = t * V_DIM + ncol;                   // w-tile 0
                    uint4 p;
                    p.x = pack_bf16x2(d0[0], d0[1]);
                    p.y = pack_bf16x2(d0[2], d0[3]);
                    p.z = pack_bf16x2(d0[4], d0[5]);
                    p.w = pack_bf16x2(d0[6], d0[7]);
                    *reinterpret_cast<uint4*>(XAs + j * XA_STRIDE + ct * 16 + half * 8) = p;
                }
                int w = 16 + ncol;                              // w-tile 1
                if (w < V_DIM) {
                    int j = t * V_DIM + w;
                    uint4 p;
                    p.x = pack_bf16x2(d1[0], d1[1]);
                    p.y = pack_bf16x2(d1[2], d1[3]);
                    p.z = pack_bf16x2(d1[4], d1[5]);
                    p.w = pack_bf16x2(d1[6], d1[7]);
                    *reinterpret_cast<uint4*>(XAs + j * XA_STRIDE + ct * 16 + half * 8) = p;
                }
            }
        }
        __syncthreads();

        // ---- stage B: acc += Wc[s] @ XA (K=64 = 2 x K32 WMMAs/tile), ----
        // ---- B fragments software-pipelined one tile ahead           ----
        v16bf a0 = load_afrag(Ws + (ow * 16) * WS_STRIDE + 0,  WS_STRIDE, lane);
        v16bf a1 = load_afrag(Ws + (ow * 16) * WS_STRIDE + 32, WS_STRIDE, lane);
        v16bf b0 = load_bfrag(XAs + jbase * 16 * XA_STRIDE + 0,  XA_STRIDE, lane);
        v16bf b1 = load_bfrag(XAs + jbase * 16 * XA_STRIDE + 32, XA_STRIDE, lane);
#pragma unroll
        for (int i = 0; i < 13; ++i) {
            v16bf nb0 = b0, nb1 = b1;
            if (i < 12) {
                const unsigned short* bb = XAs + (jbase + i + 1) * 16 * XA_STRIDE;
                nb0 = load_bfrag(bb + 0,  XA_STRIDE, lane);
                nb1 = load_bfrag(bb + 32, XA_STRIDE, lane);
            }
            acc[i] = __builtin_amdgcn_wmma_f32_16x16x32_bf16(false, a0, false, b0,
                                                             (short)0, acc[i], false, false);
            acc[i] = __builtin_amdgcn_wmma_f32_16x16x32_bf16(false, a1, false, b1,
                                                             (short)0, acc[i], false, false);
            b0 = nb0; b1 = nb1;
        }
    }

    // ---- epilogue: BN(+bias) in sc/sh, residual, ReLU, store ----
    {
        const int half = lane >> 4, ncol = lane & 15;
        const size_t nbase = (size_t)n * C_DIM * T_DIM * V_DIM + (size_t)t0 * V_DIM;
#pragma unroll
        for (int i = 0; i < 13; ++i) {
            int j = (jbase + i) * 16 + ncol;
            if (j < jmax) {
#pragma unroll
                for (int r = 0; r < 8; ++r) {
                    int o = ow * 16 + half * 8 + r;
                    size_t gidx = nbase + (size_t)o * (T_DIM * V_DIM) + j;
                    float val = acc[i][r] * sc[o] + sh[o] + x[gidx];
                    out[gidx] = fmaxf(val, 0.f);
                }
            }
        }
    }
}

extern "C" void kernel_launch(void* const* d_in, const int* in_sizes, int n_in,
                              void* d_out, int out_size, void* d_ws, size_t ws_size,
                              hipStream_t stream) {
    (void)in_sizes; (void)n_in; (void)out_size; (void)d_ws; (void)ws_size;
    const float* x     = (const float*)d_in[0];
    const float* A     = (const float*)d_in[1];
    const float* PA1   = (const float*)d_in[2];
    const float* PA2   = (const float*)d_in[3];
    const float* Wc    = (const float*)d_in[4];
    const float* bc    = (const float*)d_in[5];
    const float* gamma = (const float*)d_in[6];
    const float* beta  = (const float*)d_in[7];
    const float* rmean = (const float*)d_in[8];
    const float* rvar  = (const float*)d_in[9];
    float* out = (float*)d_out;

    dim3 grid(NCHUNK, 64, 1);   // (t-chunk, n)
    dim3 block(256, 1, 1);      // 8 wave32
    unit_gcn_fused_wmma<<<grid, block, SMEM_BYTES, stream>>>(
        x, A, PA1, PA2, Wc, bc, gamma, beta, rmean, rvar, out);
}
